// Submodular2Comm_13932873908846
// MI455X (gfx1250) — matrix operand order repaired and checked by reading, not compiled, not run on hardware
//
#include <hip/hip_runtime.h>
#include <hip/hip_bf16.h>

#define HWp  32768
#define Wd   256
#define Hd   128
#define NB   4
#define NAg  5
#define NA   20
#define NC   64
#define EPSF 1e-6f

typedef __attribute__((ext_vector_type(16))) __bf16 v16bf;
typedef __attribute__((ext_vector_type(8)))  float  v8f;

// ---------- helpers ----------
__device__ __forceinline__ float bf2f(unsigned short u) {
    return __uint_as_float(((unsigned)u) << 16);
}
__device__ __forceinline__ unsigned short f2bf(float f) {
    unsigned u = __float_as_uint(f);
    unsigned r = 0x7FFFu + ((u >> 16) & 1u);
    return (unsigned short)((u + r) >> 16);
}
__device__ __forceinline__ float clip01(float v) {
    return fminf(fmaxf(v, 0.0f), 1.0f);
}

struct Samp { int x0, y0; float w00, w01, w10, w11; };

__device__ __forceinline__ Samp make_samp(const float* M6, int h, int w) {
    float xs = 2.0f * ((float)w + 0.5f) * (1.0f / (float)Wd) - 1.0f;
    float ys = 2.0f * ((float)h + 0.5f) * (1.0f / (float)Hd) - 1.0f;
    float gx = M6[0] * xs + M6[1] * ys + M6[2];
    float gy = M6[3] * xs + M6[4] * ys + M6[5];
    float px = ((gx + 1.0f) * (float)Wd - 1.0f) * 0.5f;
    float py = ((gy + 1.0f) * (float)Hd - 1.0f) * 0.5f;
    float x0f = floorf(px), y0f = floorf(py);
    Samp s;
    s.x0 = (int)x0f; s.y0 = (int)y0f;
    float fx = px - x0f, fy = py - y0f;
    s.w00 = (1.0f - fy) * (1.0f - fx);
    s.w01 = (1.0f - fy) * fx;
    s.w10 = fy * (1.0f - fx);
    s.w11 = fy * fx;
    return s;
}
__device__ __forceinline__ float corner(const float* plane, int y, int x) {
    if ((unsigned)y < (unsigned)Hd && (unsigned)x < (unsigned)Wd)
        return plane[y * Wd + x];
    return 0.0f;
}
__device__ __forceinline__ const float* mats_for(const float* mats, int nf) {
    // normalized_affine_matrix[b, 0, n] with b = nf/5, n = nf%5
    return mats + ((size_t)(nf / NAg) * (NAg * NAg) + (nf % NAg)) * 6;
}

// ---------- kernels ----------
__global__ void k_init(float* eff) {
    if (threadIdx.x == 0 && blockIdx.x == 0) eff[0] = 0.0f;
}

__global__ void k_warp_bf16(const float* __restrict__ src, const float* __restrict__ mats,
                            unsigned short* __restrict__ dst) {
    int nf = blockIdx.y;
    int idx = blockIdx.x * 256 + threadIdx.x;
    if (idx >= NC * HWp) return;
    int c = idx / HWp, p = idx - c * HWp, h = p >> 8, w = p & 255;
    Samp s = make_samp(mats_for(mats, nf), h, w);
    const float* plane = src + ((size_t)nf * NC + c) * HWp;
    float v = corner(plane, s.y0, s.x0) * s.w00 + corner(plane, s.y0, s.x0 + 1) * s.w01 +
              corner(plane, s.y0 + 1, s.x0) * s.w10 + corner(plane, s.y0 + 1, s.x0 + 1) * s.w11;
    dst[((size_t)nf * NC + c) * HWp + p] = f2bf(v);
}

__global__ void k_warp_f32(const float* __restrict__ src, const float* __restrict__ mats,
                           float* __restrict__ dst, int C) {
    int nf = blockIdx.y;
    int idx = blockIdx.x * 256 + threadIdx.x;
    if (idx >= C * HWp) return;
    int c = idx / HWp, p = idx - c * HWp, h = p >> 8, w = p & 255;
    Samp s = make_samp(mats_for(mats, nf), h, w);
    const float* plane = src + ((size_t)nf * C + c) * HWp;
    float v = corner(plane, s.y0, s.x0) * s.w00 + corner(plane, s.y0, s.x0 + 1) * s.w01 +
              corner(plane, s.y0 + 1, s.x0) * s.w10 + corner(plane, s.y0 + 1, s.x0 + 1) * s.w11;
    dst[((size_t)nf * C + c) * HWp + p] = v;
}

__global__ void k_warp_vis(const float* __restrict__ mats, float* __restrict__ vis) {
    int nf = blockIdx.y;
    int p = blockIdx.x * 256 + threadIdx.x;
    if (p >= HWp) return;
    int h = p >> 8, w = p & 255;
    Samp s = make_samp(mats_for(mats, nf), h, w);
    float v = 0.0f;
    if ((unsigned)s.y0 < (unsigned)Hd && (unsigned)s.x0 < (unsigned)Wd) v += s.w00;
    if ((unsigned)s.y0 < (unsigned)Hd && (unsigned)(s.x0 + 1) < (unsigned)Wd) v += s.w01;
    if ((unsigned)(s.y0 + 1) < (unsigned)Hd && (unsigned)s.x0 < (unsigned)Wd) v += s.w10;
    if ((unsigned)(s.y0 + 1) < (unsigned)Hd && (unsigned)(s.x0 + 1) < (unsigned)Wd) v += s.w11;
    vis[(size_t)nf * HWp + p] = v;
}

// energy[p] = (1/64) * sum_c wx[c,p]^2 via diag(X X^T) with two accumulating bf16 WMMAs
__global__ void k_energy_wmma(const unsigned short* __restrict__ wx, float* __restrict__ energy) {
    __shared__ unsigned short tile[8][16][66];
    const int n = blockIdx.y;
    const int wid = threadIdx.x >> 5;
    const int lane = threadIdx.x & 31;
    const int pbase = blockIdx.x * 128 + wid * 16;

    // vectorized fill: each lane moves 4x 16B chunks (8 pixels of one channel each),
    // transposing into a [pixel][channel] LDS tile
    const unsigned short* base = wx + (size_t)n * NC * HWp + pbase;
    #pragma unroll
    for (int i = 0; i < 4; ++i) {
        int cid = i * 32 + lane;          // 0..127 : chunk = (channel, pixel-half)
        int c = cid >> 1, ph = (cid & 1) * 8;
        const uint4 v = *(const uint4*)(base + (size_t)c * HWp + ph);
        unsigned d[4] = {v.x, v.y, v.z, v.w};
        #pragma unroll
        for (int k = 0; k < 4; ++k) {
            tile[wid][ph + 2 * k][c]     = (unsigned short)(d[k] & 0xFFFFu);
            tile[wid][ph + 2 * k + 1][c] = (unsigned short)(d[k] >> 16);
        }
    }
    __syncthreads();

    v8f acc = {0.f, 0.f, 0.f, 0.f, 0.f, 0.f, 0.f, 0.f};
    const int m = lane & 15;
    const int half = lane >> 4;
    #pragma unroll
    for (int ko = 0; ko < 64; ko += 32) {
        v16bf Av, Bv;
        #pragma unroll
        for (int v = 0; v < 8; ++v) {
            #pragma unroll
            for (int s = 0; s < 2; ++s) {
                // A-matrix 16x32 bf16 layout: lane m; half selects K {0..7,16..23} vs {8..15,24..31}
                int ka = (v < 4) ? (2 * v + s + 8 * half) : (16 + 2 * (v - 4) + s + 8 * half);
                // B-matrix 32x16 bf16 layout: lane = column; half selects K {0..7,16..23} vs {8..15,24..31}
                int kb = ka;
                unsigned short ua = tile[wid][m][ko + ka];
                unsigned short ub = tile[wid][m][ko + kb];
                Av[v * 2 + s] = __builtin_bit_cast(__bf16, ua);
                Bv[v * 2 + s] = __builtin_bit_cast(__bf16, ub);
            }
        }
        acc = __builtin_amdgcn_wmma_f32_16x16x32_bf16(false, Av, false, Bv, (short)0, acc,
                                                      false, false);
    }
    float af[8];
    #pragma unroll
    for (int i = 0; i < 8; ++i) af[i] = acc[i];
    // diagonal of 16x16 f32 D: lanes 0..7 -> pixel=lane at VGPR lane; lanes 24..31 -> pixel=lane-16
    int pix = -1; float e = 0.0f;
    if (lane < 8)       { pix = lane;      e = af[lane]; }
    else if (lane >= 24){ pix = lane - 16; e = af[lane - 24]; }
    if (pix >= 0) energy[(size_t)n * HWp + pbase + pix] = e * (1.0f / 64.0f);
}

__global__ void k_maps(const float* __restrict__ conf, const float* __restrict__ vis,
                       float* __restrict__ scraw, float* __restrict__ sfull,
                       float* __restrict__ commf) {
    int idx = blockIdx.x * 256 + threadIdx.x;
    if (idx >= NA * HWp) return;
    int n = idx / HWp, p = idx - n * HWp;
    float c0 = conf[((size_t)n * 2 + 0) * HWp + p];
    float c1 = conf[((size_t)n * 2 + 1) * HWp + p];
    float s0 = 1.0f / (1.0f + __expf(-c0));
    float s1 = 1.0f / (1.0f + __expf(-c1));
    float sr = fmaxf(s0, s1);
    float visf = vis[idx] > 0.0f ? 1.0f : 0.0f;
    scraw[idx] = sr;
    sfull[idx] = 0.5f * (clip01(c0) + clip01(c1)) * visf;
    commf[idx] = ((n % NAg) == 0) ? 1.0f : (sr > 0.5f ? 1.0f : 0.0f);
}

__device__ __forceinline__ void img_select(int img, const float* energy, const float* scraw,
                                           const float* praw, const float* vis,
                                           const float** map, const float** msk) {
    if (img < 20)      { *map = energy + (size_t)img * HWp;        *msk = vis + (size_t)img * HWp; }
    else if (img < 40) { *map = scraw + (size_t)(img - 20) * HWp;  *msk = vis + (size_t)(img - 20) * HWp; }
    else               { *map = praw + (size_t)(img - 40) * HWp;   *msk = vis + (size_t)(img - 40) * NAg * HWp; }
}

__global__ void k_minmax(const float* __restrict__ energy, const float* __restrict__ scraw,
                         const float* __restrict__ praw, const float* __restrict__ vis,
                         unsigned* __restrict__ stats, int img_base) {
    int img = img_base + blockIdx.y;
    const float *map, *msk;
    img_select(img, energy, scraw, praw, vis, &map, &msk);
    unsigned lmin = 0x7F800000u, lmax = 0u;  // maps are all non-negative
    for (int p = threadIdx.x; p < HWp; p += 256) {
        if (msk[p] > 0.0f) {
            unsigned u = __float_as_uint(map[p]);
            lmin = min(lmin, u);
            lmax = max(lmax, u);
        }
    }
    __shared__ unsigned smn[256], smx[256];
    smn[threadIdx.x] = lmin; smx[threadIdx.x] = lmax;
    __syncthreads();
    for (int s = 128; s > 0; s >>= 1) {
        if (threadIdx.x < s) {
            smn[threadIdx.x] = min(smn[threadIdx.x], smn[threadIdx.x + s]);
            smx[threadIdx.x] = max(smx[threadIdx.x], smx[threadIdx.x + s]);
        }
        __syncthreads();
    }
    if (threadIdx.x == 0) { stats[img * 2] = smn[0]; stats[img * 2 + 1] = smx[0]; }
}

__global__ void k_hist(const float* __restrict__ energy, const float* __restrict__ scraw,
                       const float* __restrict__ praw, const float* __restrict__ vis,
                       const unsigned* __restrict__ stats, int* __restrict__ hist,
                       int img_base) {
    int img = img_base + blockIdx.y;
    const float *map, *msk;
    img_select(img, energy, scraw, praw, vis, &map, &msk);
    float mn = __uint_as_float(stats[img * 2]);
    float mx = __uint_as_float(stats[img * 2 + 1]);
    float scale = 255.0f / (mx - mn + EPSF);
    __shared__ int sh[256];
    sh[threadIdx.x] = 0;
    __syncthreads();
    for (int p = threadIdx.x; p < HWp; p += 256) {
        if (msk[p] > 0.0f) {
            int b = (int)((map[p] - mn) * scale);
            b = b < 0 ? 0 : (b > 255 ? 255 : b);
            atomicAdd(&sh[b], 1);
        }
    }
    __syncthreads();
    hist[img * 256 + threadIdx.x] = sh[threadIdx.x];
}

__device__ float hquant(const int* hb, float mn, float mx, int cnt, float q) {
    float pos = q * (float)(cnt - 1 > 0 ? cnt - 1 : 0);
    int cum = 0;
    for (int i = 0; i < 256; ++i) {
        int c = hb[i];
        if ((float)(cum + c) > pos || i == 255) {
            float frac = c > 0 ? (pos - (float)cum) / (float)c : 0.0f;
            frac = clip01(frac);
            return mn + ((float)i + frac) * (1.0f / 255.0f) * (mx - mn);
        }
        cum += c;
    }
    return mx;
}

__global__ void k_quant(const int* __restrict__ hist, const unsigned* __restrict__ stats,
                        float* __restrict__ qp, int img_base, int count) {
    int t = blockIdx.x * blockDim.x + threadIdx.x;
    if (t >= count) return;
    int img = img_base + t;
    const int* hb = hist + img * 256;
    float mn = __uint_as_float(stats[img * 2]);
    float mx = __uint_as_float(stats[img * 2 + 1]);
    int cnt = 0;
    for (int i = 0; i < 256; ++i) cnt += hb[i];
    float lo = 0.0f, hi = 0.0f;
    if (cnt > 0 && mx >= mn) {
        lo = hquant(hb, mn, mx, cnt, 0.2f);
        hi = hquant(hb, mn, mx, cnt, 0.8f);
    }
    qp[img * 2] = lo;
    qp[img * 2 + 1] = hi;
}

__global__ void k_pattern(const float* __restrict__ energy, const float* __restrict__ scraw,
                          const float* __restrict__ vis, const float* __restrict__ qp,
                          float* __restrict__ praw) {
    int idx = blockIdx.x * 256 + threadIdx.x;
    if (idx >= NB * HWp) return;
    int b = idx / HWp, p = idx - b * HWp;
    int n0 = b * NAg;
    float vis0 = vis[(size_t)n0 * HWp + p];
    float visf0 = vis0 > 0.0f ? 1.0f : 0.0f;

    float elo0 = qp[n0 * 2], ehi0 = qp[n0 * 2 + 1];
    float clo0 = qp[(20 + n0) * 2], chi0 = qp[(20 + n0) * 2 + 1];
    float e0 = energy[(size_t)n0 * HWp + p];
    float c0 = scraw[(size_t)n0 * HWp + p];
    float se0 = clip01((e0 - elo0) / (ehi0 - elo0 + EPSF)) * visf0;
    float sc0 = clip01((c0 - clo0) / (chi0 - clo0 + EPSF)) * visf0;
    // qnorm01(1 - S) via quantile equivariance: q.2(S)=0, q.8(S)=(hi-lo)/(hi-lo+eps)
    float g8e = clip01((ehi0 - elo0) / (ehi0 - elo0 + EPSF));
    float g8c = clip01((chi0 - clo0) / (chi0 - clo0 + EPSF));
    float loe = 1.0f - g8e, loc = 1.0f - g8c;
    float De = clip01(((1.0f - se0) - loe) / ((1.0f - loe) + EPSF)) * visf0;
    float Dc = clip01(((1.0f - sc0) - loc) / ((1.0f - loc) + EPSF)) * visf0;
    float D = (0.7f * De + 0.3f * Dc) * visf0;
    // q20 of S_e0 / S_c0 collapses to 0 -> K sets are (S<=0) within mask
    float Kempty = ((se0 <= 0.0f) && (sc0 <= 0.0f) && (vis0 > 0.0f)) ? 1.0f : 0.0f;

    float Sup = 0.0f;
    for (int j = 1; j < NAg; ++j) {
        int n = n0 + j;
        float vf = vis[(size_t)n * HWp + p] > 0.0f ? 1.0f : 0.0f;
        float elo = qp[n * 2], ehi = qp[n * 2 + 1];
        float clo = qp[(20 + n) * 2], chi = qp[(20 + n) * 2 + 1];
        float sen = clip01((energy[(size_t)n * HWp + p] - elo) / (ehi - elo + EPSF)) * vf;
        float scn = clip01((scraw[(size_t)n * HWp + p] - clo) / (chi - clo + EPSF)) * vf;
        Sup += (0.8f * sen + 0.2f * scn) * vf;
    }
    Sup *= 0.25f;
    praw[(size_t)b * HWp + p] = D * Sup * (1.0f - Kempty);
}

__global__ void k_scores(const float* __restrict__ praw, const float* __restrict__ qp,
                         const float* __restrict__ sfull, const float* __restrict__ vis,
                         const float* __restrict__ commf, float* __restrict__ owner) {
    int idx = blockIdx.x * 256 + threadIdx.x;
    if (idx >= NB * HWp) return;
    int b = idx / HWp, p = idx - b * HWp;
    int n0 = b * NAg;
    float plo = qp[(40 + b) * 2], phi = qp[(40 + b) * 2 + 1];
    float visf0 = vis[(size_t)n0 * HWp + p] > 0.0f ? 1.0f : 0.0f;
    float Pq = clip01((praw[(size_t)b * HWp + p] - plo) / (phi - plo + EPSF)) * visf0;
    float gate = 0.1f + 0.9f * Pq;
    int win = 0;
    float best = -1.0f;
    for (int j = 0; j < NAg; ++j) {
        int n = n0 + j;
        float vf = vis[(size_t)n * HWp + p] > 0.0f ? 1.0f : 0.0f;
        float sc = sfull[(size_t)n * HWp + p] * gate * vf;
        if (sc > best) { best = sc; win = j; }
    }
    for (int j = 0; j < NAg; ++j) {
        int n = n0 + j;
        owner[(size_t)n * HWp + p] =
            (j == win && commf[(size_t)n * HWp + p] > 0.0f) ? 1.0f : 0.0f;
    }
}

// One block per (batch, image row). Stage owner[5 agents][3 rows][256] into LDS with
// async global->LDS copies (ASYNCcnt), then dilate out of LDS.
__global__ void k_final(const unsigned short* __restrict__ wx, const float* __restrict__ commf,
                        const float* __restrict__ owner, float* __restrict__ out,
                        float* __restrict__ eff) {
    __shared__ float sh[NAg * 3 * Wd];  // 15KB
    const int blk = blockIdx.x;         // 0 .. NB*Hd-1
    const int b = blk / Hd, h = blk - b * Hd;
    const int w = threadIdx.x;          // 0..255 == column
    const int n0 = b * NAg;
    const int p = h * Wd + w;

    // stage: 5 agents * 3 rows * 64 chunks of 16B
    const int NCHUNK = NAg * 3 * (Wd / 4);
    for (int cid = threadIdx.x; cid < NCHUNK; cid += 256) {
        int j = cid / (3 * (Wd / 4));
        int rem = cid - j * 3 * (Wd / 4);
        int r = rem / (Wd / 4);
        int seg = rem - r * (Wd / 4);
        int hh = h + r - 1;
        int li = (j * 3 + r) * Wd + seg * 4;
        if ((unsigned)hh < (unsigned)Hd) {
            const float* g = owner + (size_t)(n0 + j) * HWp + hh * Wd + seg * 4;
            unsigned lds_off = (unsigned)(size_t)(const void*)&sh[li];
            asm volatile("global_load_async_to_lds_b128 %0, %1, off"
                         :: "v"(lds_off), "v"((unsigned long long)(size_t)g)
                         : "memory");
        } else {
            sh[li + 0] = 0.0f; sh[li + 1] = 0.0f; sh[li + 2] = 0.0f; sh[li + 3] = 0.0f;
        }
    }
    asm volatile("s_wait_asynccnt 0x0" ::: "memory");
    __syncthreads();

    bool fin[NAg];
    int cnt = 0;
    #pragma unroll
    for (int j = 0; j < NAg; ++j) {
        float m = 0.0f;
        #pragma unroll
        for (int r = 0; r < 3; ++r) {
            const float* row = &sh[(j * 3 + r) * Wd];
            if (w > 0)       m = fmaxf(m, row[w - 1]);
            m = fmaxf(m, row[w]);
            if (w < Wd - 1)  m = fmaxf(m, row[w + 1]);
        }
        bool f = (m > 0.0f) && (commf[(size_t)(n0 + j) * HWp + p] > 0.0f);
        fin[j] = f;
        if (j >= 1 && f) cnt++;
    }
    for (int c = 0; c < NC; ++c) {
        float best = 0.0f;
        #pragma unroll
        for (int j = 0; j < NAg; ++j) {
            float t = fin[j] ? bf2f(wx[((size_t)(n0 + j) * NC + c) * HWp + p]) : 0.0f;
            best = (j == 0) ? t : fmaxf(best, t);
        }
        out[((size_t)b * NC + c) * HWp + p] = best;
    }
    __shared__ int scnt[256];
    scnt[threadIdx.x] = cnt;
    __syncthreads();
    for (int s = 128; s > 0; s >>= 1) {
        if (threadIdx.x < s) scnt[threadIdx.x] += scnt[threadIdx.x + s];
        __syncthreads();
    }
    if (threadIdx.x == 0)
        atomicAdd(eff, (float)scnt[0] * (1.0f / ((float)(NAg - 1) * (float)HWp * (float)NB)));
}

// ---------- launcher ----------
extern "C" void kernel_launch(void* const* d_in, const int* in_sizes, int n_in,
                              void* d_out, int out_size, void* d_ws, size_t ws_size,
                              hipStream_t stream) {
    const float* x = (const float*)d_in[0];
    const float* psm = (const float*)d_in[1];
    const float* mats = (const float*)d_in[3];

    char* ws = (char*)d_ws;
    size_t off = 0;
    unsigned short* wx = (unsigned short*)(ws + off); off += (size_t)NA * NC * HWp * sizeof(unsigned short);
    float* vis    = (float*)(ws + off); off += (size_t)NA * HWp * sizeof(float);
    float* conf   = (float*)(ws + off); off += (size_t)NA * 2 * HWp * sizeof(float);
    float* energy = (float*)(ws + off); off += (size_t)NA * HWp * sizeof(float);
    float* scraw  = (float*)(ws + off); off += (size_t)NA * HWp * sizeof(float);
    float* sfull  = (float*)(ws + off); off += (size_t)NA * HWp * sizeof(float);
    float* commf  = (float*)(ws + off); off += (size_t)NA * HWp * sizeof(float);
    float* owner  = (float*)(ws + off); off += (size_t)NA * HWp * sizeof(float);
    float* praw   = (float*)(ws + off); off += (size_t)NB * HWp * sizeof(float);
    float* qp     = (float*)(ws + off); off += 128 * sizeof(float);
    unsigned* stats = (unsigned*)(ws + off); off += 128 * sizeof(unsigned);
    int* hist     = (int*)(ws + off); off += 44 * 256 * sizeof(int);

    float* out = (float*)d_out;
    float* eff = out + (size_t)NB * NC * HWp;

    dim3 blk(256);
    k_init<<<1, 64, 0, stream>>>(eff);
    k_warp_bf16<<<dim3((NC * HWp) / 256, NA), blk, 0, stream>>>(x, mats, wx);
    k_warp_f32<<<dim3((2 * HWp) / 256, NA), blk, 0, stream>>>(psm, mats, conf, 2);
    k_warp_vis<<<dim3(HWp / 256, NA), blk, 0, stream>>>(mats, vis);
    k_energy_wmma<<<dim3(HWp / 128, NA), blk, 0, stream>>>(wx, energy);
    k_maps<<<dim3((NA * HWp) / 256), blk, 0, stream>>>(conf, vis, scraw, sfull, commf);
    k_minmax<<<dim3(1, 40), blk, 0, stream>>>(energy, scraw, praw, vis, stats, 0);
    k_hist<<<dim3(1, 40), blk, 0, stream>>>(energy, scraw, praw, vis, stats, hist, 0);
    k_quant<<<1, 64, 0, stream>>>(hist, stats, qp, 0, 40);
    k_pattern<<<dim3((NB * HWp) / 256), blk, 0, stream>>>(energy, scraw, vis, qp, praw);
    k_minmax<<<dim3(1, NB), blk, 0, stream>>>(energy, scraw, praw, vis, stats, 40);
    k_hist<<<dim3(1, NB), blk, 0, stream>>>(energy, scraw, praw, vis, stats, hist, 40);
    k_quant<<<1, 8, 0, stream>>>(hist, stats, qp, 40, NB);
    k_scores<<<dim3((NB * HWp) / 256), blk, 0, stream>>>(praw, qp, sfull, vis, commf, owner);
    k_final<<<dim3(NB * Hd), blk, 0, stream>>>(wx, commf, owner, out, eff);
}
